// MultiHeadAttentionSimple_11639361372684
// MI455X (gfx1250) — matrix operand order repaired and checked by reading, not compiled, not run on hardware
//
#include <hip/hip_runtime.h>
#include <stdint.h>

// MI455X / gfx1250, wave32. bf16 WMMA (v_wmma_f32_16x16x32_bf16) everywhere.
// GEMM staging uses GLOBAL_LOAD_ASYNC_TO_LDS_B128 + ASYNCcnt double buffering
// when the toolchain exposes the builtins; otherwise sync ds_store staging.

typedef __attribute__((ext_vector_type(16))) __bf16 v16bf;
typedef __attribute__((ext_vector_type(8)))  __bf16 v8bf;
typedef __attribute__((ext_vector_type(8)))  float  v8f;
typedef __attribute__((__vector_size__(4 * sizeof(int)))) int v4i;

#define BQ   4
#define SQ   2048
#define EQ   1024
#define HQ   16
#define DHQ  64
#define MROWS (BQ*SQ)          // 8192

#if defined(__has_builtin)
#  if __has_builtin(__builtin_amdgcn_global_load_async_to_lds_b128) && \
      __has_builtin(__builtin_amdgcn_s_wait_asynccnt)
#    define USE_ASYNC 1
#  endif
#endif
#ifndef USE_ASYNC
#  define USE_ASYNC 0
#endif

#if USE_ASYNC
typedef __attribute__((address_space(1))) v4i* gp128_t;   // global (device) v4i*
typedef __attribute__((address_space(3))) v4i* lp128_t;   // LDS (shared) v4i*

__device__ __forceinline__ gp128_t as_gp128(const void* p) {
    return (gp128_t)(v4i*)const_cast<void*>(p);
}
__device__ __forceinline__ lp128_t as_lp128(void* p) {
    return (lp128_t)(v4i*)p;
}
#endif

__device__ __forceinline__ uint16_t f2bf(float f) {
    union { float f; uint32_t u; } x; x.f = f;
    uint32_t r = (x.u + 0x7FFFu + ((x.u >> 16) & 1u)) >> 16;   // RNE
    return (uint16_t)r;
}

__device__ __forceinline__ v16bf cat16(v8bf lo, v8bf hi) {
    return __builtin_shufflevector(lo, hi, 0,1,2,3,4,5,6,7,8,9,10,11,12,13,14,15);
}

// ---------------- prep kernels: fp32 -> bf16 packing ----------------

__global__ void k_cvt_bf16(const float* __restrict__ src, uint16_t* __restrict__ dst, int n) {
    int i = blockIdx.x * 256 + threadIdx.x;
    if (i < n) dst[i] = f2bf(src[i]);
}

// Wq/Wk/Wv are [H,E,DH]; pack to Bt[3*1024][1024] bf16, row n = (p,h,d), col = e.
__global__ void k_pack_qkvw(const float* __restrict__ Wq, const float* __restrict__ Wk,
                            const float* __restrict__ Wv, uint16_t* __restrict__ wt) {
    int i = blockIdx.x * 256 + threadIdx.x;
    if (i >= 3 * EQ * EQ) return;
    int p = i >> 20;
    int r = i & (EQ * EQ - 1);
    int n = r >> 10, e = r & 1023;
    int h = n >> 6, d = n & 63;
    const float* W = (p == 0) ? Wq : (p == 1) ? Wk : Wv;
    wt[i] = f2bf(W[((size_t)h * EQ + e) * DHQ + d]);
}

// Wp [E,E] -> WpT[n][e] bf16
__global__ void k_pack_wp(const float* __restrict__ Wp, uint16_t* __restrict__ wpt) {
    int i = blockIdx.x * 256 + threadIdx.x;
    if (i >= EQ * EQ) return;
    int n = i >> 10, e = i & 1023;
    wpt[i] = f2bf(Wp[(size_t)e * EQ + n]);
}

// ---------------- shared WMMA GEMM core ----------------
// C[M0..+128, N0..+128] = A[M,K]bf16 (row-major) x Bt[N,K]bf16 (row-major, i.e. B^T)
// 256 threads = 8 waves, wave tile 64x32 = 4x2 16x16 accumulators.
// LDS rows padded to 48 elems (96B) so every 16B chunk stays 16B-aligned.

__device__ void gemm_tiles_bf16(const uint16_t* __restrict__ A,
                                const uint16_t* __restrict__ Bt,
                                int K, int M0, int N0, v8f acc[4][2]) {
#if USE_ASYNC
    __shared__ __align__(16) uint16_t sA[2][128 * 48];
    __shared__ __align__(16) uint16_t sB[2][128 * 48];
#else
    __shared__ __align__(16) uint16_t sA[1][128 * 48];
    __shared__ __align__(16) uint16_t sB[1][128 * 48];
#endif
    const int t    = threadIdx.x;
    const int w    = t >> 5;
    const int lane = t & 31;
    const int moff = (w >> 2) * 64;
    const int noff = (w & 3) * 32;
    const int l16  = lane & 15;
    const int kbA  = (lane < 16) ? 0 : 16;   // A lane: K 0-15 / 16-31 contiguous
    const int kbB  = (lane < 16) ? 0 : 16;   // B lane: K 0-15 / 16-31 contiguous

    const v8f vz = {0.f,0.f,0.f,0.f,0.f,0.f,0.f,0.f};
    #pragma unroll
    for (int i = 0; i < 4; ++i)
        #pragma unroll
        for (int j = 0; j < 2; ++j) acc[i][j] = vz;

    const int row0 = t >> 2,          col8 = (t & 3) * 8;          // chunk c = t
    const int row1 = (t + 256) >> 2;                               // chunk c = t+256 (same col8)

#if USE_ASYNC
    // ---- issue k-slice 0 into buffer 0 ----
    {
        __builtin_amdgcn_global_load_async_to_lds_b128(
            as_gp128(&A[(size_t)(M0 + row0) * K + col8]),
            as_lp128(&sA[0][row0 * 48 + col8]), 0, 0);
        __builtin_amdgcn_global_load_async_to_lds_b128(
            as_gp128(&Bt[(size_t)(N0 + row0) * K + col8]),
            as_lp128(&sB[0][row0 * 48 + col8]), 0, 0);
        __builtin_amdgcn_global_load_async_to_lds_b128(
            as_gp128(&A[(size_t)(M0 + row1) * K + col8]),
            as_lp128(&sA[0][row1 * 48 + col8]), 0, 0);
        __builtin_amdgcn_global_load_async_to_lds_b128(
            as_gp128(&Bt[(size_t)(N0 + row1) * K + col8]),
            as_lp128(&sB[0][row1 * 48 + col8]), 0, 0);
    }
    for (int kk = 0; kk < K; kk += 32) {
        const int cur = (kk >> 5) & 1;
        const int nxt = cur ^ 1;
        if (kk + 32 < K) {
            const int kn = kk + 32;
            __builtin_amdgcn_global_load_async_to_lds_b128(
                as_gp128(&A[(size_t)(M0 + row0) * K + kn + col8]),
                as_lp128(&sA[nxt][row0 * 48 + col8]), 0, 0);
            __builtin_amdgcn_global_load_async_to_lds_b128(
                as_gp128(&Bt[(size_t)(N0 + row0) * K + kn + col8]),
                as_lp128(&sB[nxt][row0 * 48 + col8]), 0, 0);
            __builtin_amdgcn_global_load_async_to_lds_b128(
                as_gp128(&A[(size_t)(M0 + row1) * K + kn + col8]),
                as_lp128(&sA[nxt][row1 * 48 + col8]), 0, 0);
            __builtin_amdgcn_global_load_async_to_lds_b128(
                as_gp128(&Bt[(size_t)(N0 + row1) * K + kn + col8]),
                as_lp128(&sB[nxt][row1 * 48 + col8]), 0, 0);
            __builtin_amdgcn_s_wait_asynccnt(4);   // current slice resident, next in flight
        } else {
            __builtin_amdgcn_s_wait_asynccnt(0);
        }
        __syncthreads();
        v16bf a[4], b[2];
        #pragma unroll
        for (int i = 0; i < 4; ++i) {
            int r = moff + i * 16 + l16;
            a[i] = cat16(*(const v8bf*)&sA[cur][r * 48 + kbA],
                         *(const v8bf*)&sA[cur][r * 48 + kbA + 8]);
        }
        #pragma unroll
        for (int j = 0; j < 2; ++j) {
            int r = noff + j * 16 + l16;
            b[j] = cat16(*(const v8bf*)&sB[cur][r * 48 + kbB],
                         *(const v8bf*)&sB[cur][r * 48 + kbB + 8]);
        }
        #pragma unroll
        for (int i = 0; i < 4; ++i)
            #pragma unroll
            for (int j = 0; j < 2; ++j)
                acc[i][j] = __builtin_amdgcn_wmma_f32_16x16x32_bf16(
                    false, a[i], false, b[j], (short)0, acc[i][j], false, false);
        __syncthreads();   // all waves done reading 'cur' before it is overwritten
    }
#else
    for (int kk = 0; kk < K; kk += 32) {
        *(v8bf*)&sA[0][row0 * 48 + col8] = *(const v8bf*)&A [(size_t)(M0 + row0) * K + kk + col8];
        *(v8bf*)&sB[0][row0 * 48 + col8] = *(const v8bf*)&Bt[(size_t)(N0 + row0) * K + kk + col8];
        *(v8bf*)&sA[0][row1 * 48 + col8] = *(const v8bf*)&A [(size_t)(M0 + row1) * K + kk + col8];
        *(v8bf*)&sB[0][row1 * 48 + col8] = *(const v8bf*)&Bt[(size_t)(N0 + row1) * K + kk + col8];
        __syncthreads();
        v16bf a[4], b[2];
        #pragma unroll
        for (int i = 0; i < 4; ++i) {
            int r = moff + i * 16 + l16;
            a[i] = cat16(*(const v8bf*)&sA[0][r * 48 + kbA],
                         *(const v8bf*)&sA[0][r * 48 + kbA + 8]);
        }
        #pragma unroll
        for (int j = 0; j < 2; ++j) {
            int r = noff + j * 16 + l16;
            b[j] = cat16(*(const v8bf*)&sB[0][r * 48 + kbB],
                         *(const v8bf*)&sB[0][r * 48 + kbB + 8]);
        }
        #pragma unroll
        for (int i = 0; i < 4; ++i)
            #pragma unroll
            for (int j = 0; j < 2; ++j)
                acc[i][j] = __builtin_amdgcn_wmma_f32_16x16x32_bf16(
                    false, a[i], false, b[j], (short)0, acc[i][j], false, false);
        __syncthreads();
    }
#endif
}

// ---------------- fused QKV projection ----------------
// N = 3072 (q|k|v). Epilogue scatters: q,k -> [B,H,S,DH] bf16; v -> [B,H,DH,S] bf16 (transposed).

__global__ __launch_bounds__(256) void k_gemm_qkv(
    const uint16_t* __restrict__ xb, const uint16_t* __restrict__ wt,
    const float* __restrict__ bq, const float* __restrict__ bk, const float* __restrict__ bv,
    uint16_t* __restrict__ qo, uint16_t* __restrict__ ko, uint16_t* __restrict__ vto) {
    v8f acc[4][2];
    const int M0 = blockIdx.y * 128, N0 = blockIdx.x * 128;
    gemm_tiles_bf16(xb, wt, EQ, M0, N0, acc);

    const int lane = threadIdx.x & 31, w = threadIdx.x >> 5;
    const int moff = (w >> 2) * 64, noff = (w & 3) * 32;
    const int l16  = lane & 15, hi8 = (lane >> 4) * 8;
    #pragma unroll
    for (int i = 0; i < 4; ++i)
        #pragma unroll
        for (int j = 0; j < 2; ++j) {
            int n   = N0 + noff + j * 16 + l16;
            int p   = n >> 10, n10 = n & 1023;
            int h   = n10 >> 6, d = n10 & 63;
            const float* bias = (p == 0) ? bq : (p == 1) ? bk : bv;
            float bval = bias[n10];
            #pragma unroll
            for (int e = 0; e < 8; ++e) {
                int m = M0 + moff + i * 16 + e + hi8;
                int b = m >> 11, s = m & 2047;       // S = 2048
                uint16_t val = f2bf(acc[i][j][e] + bval);
                size_t bh = (size_t)(b * HQ + h);
                if      (p == 0) qo [(bh * SQ + s) * DHQ + d] = val;
                else if (p == 1) ko [(bh * SQ + s) * DHQ + d] = val;
                else             vto[(bh * DHQ + d) * SQ + s] = val;
            }
        }
}

// ---------------- output projection: out = y @ Wp + bp (fp32 out) ----------------

__global__ __launch_bounds__(256) void k_gemm_out(
    const uint16_t* __restrict__ yb, const uint16_t* __restrict__ wpt,
    const float* __restrict__ bp, float* __restrict__ out) {
    v8f acc[4][2];
    const int M0 = blockIdx.y * 128, N0 = blockIdx.x * 128;
    gemm_tiles_bf16(yb, wpt, EQ, M0, N0, acc);

    const int lane = threadIdx.x & 31, w = threadIdx.x >> 5;
    const int moff = (w >> 2) * 64, noff = (w & 3) * 32;
    const int l16  = lane & 15, hi8 = (lane >> 4) * 8;
    #pragma unroll
    for (int i = 0; i < 4; ++i)
        #pragma unroll
        for (int j = 0; j < 2; ++j) {
            int n = N0 + noff + j * 16 + l16;
            float bval = bp[n];
            #pragma unroll
            for (int e = 0; e < 8; ++e) {
                int m = M0 + moff + i * 16 + e + hi8;
                out[(size_t)m * EQ + n] = acc[i][j][e] + bval;
            }
        }
}

// ---------------- flash attention: 1 wave handles 16 query rows ----------------
// scores = (Q K^T)/8 with key-padding mask, online softmax, O += P V.
// C-layout striping: VGPR e <-> rows e (lanes 0-15) and e+8 (lanes 16-31); col = lane%16.

__global__ __launch_bounds__(32) void k_attn(
    const uint16_t* __restrict__ q, const uint16_t* __restrict__ k,
    const uint16_t* __restrict__ vt, const unsigned char* __restrict__ mask,
    uint16_t* __restrict__ y) {
    __shared__ __align__(16) uint16_t sP[16 * 32];

    const int bid   = blockIdx.x;
    const int qt    = bid & (SQ / 16 - 1);   // 128 q-tiles
    const int bh    = bid >> 7;
    const int h     = bh & (HQ - 1);
    const int b     = bh >> 4;
    const int qbase = qt * 16;
    const int lane  = threadIdx.x;
    const int l16   = lane & 15;
    const int hi8   = (lane >> 4) * 8;
    const int kbA   = (lane < 16) ? 0 : 16;
    const int kbB   = (lane < 16) ? 0 : 16;

    const uint16_t* qp = q  + ((size_t)bh * SQ + qbase) * DHQ;
    const uint16_t* kp = k  + (size_t)bh * SQ * DHQ;
    const uint16_t* vp = vt + (size_t)bh * DHQ * SQ;
    const unsigned char* mp = mask + (size_t)b * SQ;

    // Q A-operands (K-dim = 64 -> two 16x32 operands), loaded once
    const uint16_t* qrow = qp + (size_t)l16 * DHQ;
    v16bf aq0 = cat16(*(const v8bf*)&qrow[kbA],      *(const v8bf*)&qrow[kbA + 8]);
    v16bf aq1 = cat16(*(const v8bf*)&qrow[32 + kbA], *(const v8bf*)&qrow[32 + kbA + 8]);

    const v8f vz = {0.f,0.f,0.f,0.f,0.f,0.f,0.f,0.f};
    v8f o[4] = {vz, vz, vz, vz};
    float rm[8], rl[8];
    #pragma unroll
    for (int e = 0; e < 8; ++e) { rm[e] = -__builtin_inff(); rl[e] = 0.f; }

    for (int kt = 0; kt < SQ; kt += 32) {
        // --- scores for two 16-key tiles ---
        const uint16_t* kr0 = kp + (size_t)(kt + l16) * DHQ;
        const uint16_t* kr1 = kp + (size_t)(kt + 16 + l16) * DHQ;
        v16bf bk0a = cat16(*(const v8bf*)&kr0[kbB],      *(const v8bf*)&kr0[kbB + 8]);
        v16bf bk0b = cat16(*(const v8bf*)&kr0[32 + kbB], *(const v8bf*)&kr0[32 + kbB + 8]);
        v16bf bk1a = cat16(*(const v8bf*)&kr1[kbB],      *(const v8bf*)&kr1[kbB + 8]);
        v16bf bk1b = cat16(*(const v8bf*)&kr1[32 + kbB], *(const v8bf*)&kr1[32 + kbB + 8]);
        v8f s0 = vz, s1 = vz;
        s0 = __builtin_amdgcn_wmma_f32_16x16x32_bf16(false, aq0, false, bk0a, (short)0, s0, false, false);
        s0 = __builtin_amdgcn_wmma_f32_16x16x32_bf16(false, aq1, false, bk0b, (short)0, s0, false, false);
        s1 = __builtin_amdgcn_wmma_f32_16x16x32_bf16(false, aq0, false, bk1a, (short)0, s1, false, false);
        s1 = __builtin_amdgcn_wmma_f32_16x16x32_bf16(false, aq1, false, bk1b, (short)0, s1, false, false);

        const bool msk0 = mp[kt + l16] != 0;        // this lane's key column
        const bool msk1 = mp[kt + 16 + l16] != 0;
        const float NINF = -__builtin_inff();

        // --- online softmax (per row; rows striped across VGPR index + lane half) ---
        #pragma unroll
        for (int e = 0; e < 8; ++e) {
            float v0 = msk0 ? NINF : s0[e] * 0.125f;
            float v1 = msk1 ? NINF : s1[e] * 0.125f;
            float tm = fmaxf(v0, v1);
            tm = fmaxf(tm, __shfl_xor(tm, 1, 16));
            tm = fmaxf(tm, __shfl_xor(tm, 2, 16));
            tm = fmaxf(tm, __shfl_xor(tm, 4, 16));
            tm = fmaxf(tm, __shfl_xor(tm, 8, 16));
            float nm   = fmaxf(fmaxf(rm[e], tm), -1e30f);   // NaN-free guard
            float corr = __expf(rm[e] - nm);
            float p0   = __expf(v0 - nm);
            float p1   = __expf(v1 - nm);
            float rs   = p0 + p1;
            rs += __shfl_xor(rs, 1, 16);
            rs += __shfl_xor(rs, 2, 16);
            rs += __shfl_xor(rs, 4, 16);
            rs += __shfl_xor(rs, 8, 16);
            rl[e] = rl[e] * corr + rs;
            rm[e] = nm;
            #pragma unroll
            for (int j = 0; j < 4; ++j) o[j][e] *= corr;
            int row = e + hi8;
            sP[row * 32 + l16]      = f2bf(p0);   // D-layout -> LDS (row-major 16x32)
            sP[row * 32 + 16 + l16] = f2bf(p1);
        }
        __syncthreads();

        // P back in A-layout, then O += P * V^T-tiles
        v16bf ap = cat16(*(const v8bf*)&sP[l16 * 32 + kbA],
                         *(const v8bf*)&sP[l16 * 32 + kbA + 8]);
        #pragma unroll
        for (int j = 0; j < 4; ++j) {
            const uint16_t* vr = vp + (size_t)(j * 16 + l16) * SQ + kt;
            v16bf bvv = cat16(*(const v8bf*)&vr[kbB], *(const v8bf*)&vr[kbB + 8]);
            o[j] = __builtin_amdgcn_wmma_f32_16x16x32_bf16(
                false, ap, false, bvv, (short)0, o[j], false, false);
        }
        __syncthreads();
    }

    // normalize + store concat-heads y [B,S,E] bf16
    #pragma unroll
    for (int e = 0; e < 8; ++e) {
        float inv = 1.0f / rl[e];
        int srow = qbase + e + hi8;
        size_t base = ((size_t)b * SQ + srow) * EQ + (size_t)h * DHQ;
        #pragma unroll
        for (int j = 0; j < 4; ++j)
            y[base + j * 16 + l16] = f2bf(o[j][e] * inv);
    }
}

// ---------------- launcher ----------------

extern "C" void kernel_launch(void* const* d_in, const int* in_sizes, int n_in,
                              void* d_out, int out_size, void* d_ws, size_t ws_size,
                              hipStream_t stream) {
    (void)in_sizes; (void)n_in; (void)out_size; (void)ws_size;
    const float*         x    = (const float*)d_in[0];
    const unsigned char* mask = (const unsigned char*)d_in[1];  // jax bool -> 1 byte
    const float* Wq = (const float*)d_in[2];
    const float* bq = (const float*)d_in[3];
    const float* Wk = (const float*)d_in[4];
    const float* bk = (const float*)d_in[5];
    const float* Wv = (const float*)d_in[6];
    const float* bv = (const float*)d_in[7];
    const float* Wp = (const float*)d_in[8];
    const float* bp = (const float*)d_in[9];
    float* out = (float*)d_out;

    char* ws = (char*)d_ws;
    uint16_t* xb  = (uint16_t*)(ws);                         // 16 MB  x as bf16
    uint16_t* wt  = (uint16_t*)(ws + (size_t)16 * 1048576);  //  6 MB  packed Wq|Wk|Wv^T
    uint16_t* wpt = (uint16_t*)(ws + (size_t)22 * 1048576);  //  2 MB  Wp^T
    uint16_t* qb  = (uint16_t*)(ws + (size_t)24 * 1048576);  // 16 MB  Q  [B,H,S,DH]
    uint16_t* kb  = (uint16_t*)(ws + (size_t)40 * 1048576);  // 16 MB  K  [B,H,S,DH]
    uint16_t* vtb = (uint16_t*)(ws + (size_t)56 * 1048576);  // 16 MB  V^T[B,H,DH,S]
    uint16_t* yb  = (uint16_t*)(ws + (size_t)72 * 1048576);  // 16 MB  attn out [B,S,E]

    k_cvt_bf16 <<<(MROWS * EQ + 255) / 256, 256, 0, stream>>>(x, xb, MROWS * EQ);
    k_pack_qkvw<<<(3 * EQ * EQ + 255) / 256, 256, 0, stream>>>(Wq, Wk, Wv, wt);
    k_pack_wp  <<<(EQ * EQ + 255) / 256, 256, 0, stream>>>(Wp, wpt);

    dim3 g1(3 * EQ / 128, MROWS / 128);   // 24 x 64
    k_gemm_qkv<<<g1, 256, 0, stream>>>(xb, wt, bq, bk, bv, qb, kb, vtb);

    k_attn<<<BQ * HQ * (SQ / 16), 32, 0, stream>>>(qb, kb, vtb, mask, yb);

    dim3 g2(EQ / 128, MROWS / 128);       // 8 x 64
    k_gemm_out<<<g2, 256, 0, stream>>>(yb, wpt, bp, out);
}
